// W8Linear_34883724378155
// MI455X (gfx1250) — matrix operand order repaired and checked by reading, not compiled, not run on hardware
//
#include <hip/hip_runtime.h>

// y = x @ (q * scale[:,None]).T + bias, computed as (x_f16 @ q_f16^T)_f32 * scale + bias.
// q in [-127,127] is exact in f16; per-row scale factored into the epilogue.
// Block tile 128x256, BK=32 double-buffered LDS (61.4KB), 8 waves, 64x64 per wave
// => 16 v_wmma_f32_16x16x32_f16 per wave per K-stage.
// Weight conversion uses the i16 -> f16 single-op path (V_CVT_F16_I16).

#define TOKENS 8192
#define D_IN   4096
#define D_OUT  11008

#define BM 128
#define BN 256
#define BK 32
#define PADH 8
#define LDT (BK + PADH)          // 40 halves per LDS row (80B, 16B-aligned)
#define NK (D_IN / BK)           // 128 K stages

typedef __attribute__((ext_vector_type(16))) _Float16 v16h;
typedef __attribute__((ext_vector_type(8)))  _Float16 h8;
typedef __attribute__((ext_vector_type(4)))  _Float16 h4;
typedef __attribute__((ext_vector_type(8)))  float    v8f;
typedef __attribute__((ext_vector_type(4)))  float    f4;
typedef __attribute__((ext_vector_type(4)))  int      i4;

union ABFrag { v16h v; h8 h[2]; };

__device__ __forceinline__ h4 cvt_f4_h4(f4 a) {
    h4 r = {(_Float16)a.x, (_Float16)a.y, (_Float16)a.z, (_Float16)a.w};
    return r;
}
// int32 in [-127,127]: low 16 bits are the exact i16 -> single V_CVT_F16_I16 per elem
__device__ __forceinline__ h4 cvt_i4_h4(i4 b) {
    h4 r = {(_Float16)(short)b.x, (_Float16)(short)b.y,
            (_Float16)(short)b.z, (_Float16)(short)b.w};
    return r;
}

__global__ __launch_bounds__(256)
void w8linear_wmma(const float* __restrict__ x,
                   const int*   __restrict__ wq,
                   const float* __restrict__ scale,
                   const float* __restrict__ bias,
                   float*       __restrict__ out) {
    __shared__ _Float16 As[2][BM][LDT];   // x tile,      f16, row-major [m][k]  (20.5 KB)
    __shared__ _Float16 Bs[2][BN][LDT];   // weight tile, f16, row-major [n][k]  (41.0 KB)

    const int tid  = threadIdx.x;
    const int lane = tid & 31;            // wave32
    const int wave = tid >> 5;            // 0..7
    const int wm   = wave & 1;            // 2 wave rows  (64 tokens each)
    const int wn   = wave >> 1;           // 4 wave cols  (64 outputs each)
    const int ln   = lane & 15;
    const int lhi  = lane >> 4;           // lane half select

    // ---- L2-friendly block swizzle: 8 M-tiles share each weight panel ----
    const int nTiles = D_OUT / BN;        // 43
    const int mTiles = TOKENS / BM;       // 64
    const int GROUP  = 8;
    int pid      = blockIdx.x;
    int group_id = pid / (GROUP * nTiles);
    int first_m  = group_id * GROUP;
    int rem      = mTiles - first_m;
    int gsz      = rem < GROUP ? rem : GROUP;
    int pm       = first_m + (pid % gsz);
    int pn       = (pid % (GROUP * nTiles)) / gsz;

    const int mBase = pm * BM;
    const int nBase = pn * BN;

    v8f acc[4][4] = {};

    // ---- prologue: stage 0 -> buffer 0 ----
    #pragma unroll
    for (int i = 0; i < 4; i++) {         // A: 128x32 = 1024 float4 / 256 thr
        int idx = tid + i * 256;
        int r = idx >> 3;                 // 8 x float4 per 32-wide row
        int c = (idx & 7) << 2;
        f4 a = *(const f4*)(x + (size_t)(mBase + r) * D_IN + c);
        *(h4*)(&As[0][r][c]) = cvt_f4_h4(a);
    }
    #pragma unroll
    for (int i = 0; i < 8; i++) {         // B: 256x32 = 2048 int4 / 256 thr
        int idx = tid + i * 256;
        int r = idx >> 3;
        int c = (idx & 7) << 2;
        i4 b = *(const i4*)(wq + (size_t)(nBase + r) * D_IN + c);
        *(h4*)(&Bs[0][r][c]) = cvt_i4_h4(b);
    }
    __syncthreads();

    // ---- main loop: double-buffered LDS, 16 WMMAs per wave per stage ----
    for (int ks = 0; ks < NK; ks++) {
        const int  buf = ks & 1;
        const bool pf  = (ks + 1) < NK;

        f4 ar[4];
        i4 br[8];
        if (pf) {
            const int kn = (ks + 1) * BK;
            #pragma unroll
            for (int i = 0; i < 4; i++) {
                int idx = tid + i * 256;
                int r = idx >> 3;
                int c = (idx & 7) << 2;
                ar[i] = *(const f4*)(x + (size_t)(mBase + r) * D_IN + kn + c);
            }
            #pragma unroll
            for (int i = 0; i < 8; i++) {
                int idx = tid + i * 256;
                int r = idx >> 3;
                int c = (idx & 7) << 2;
                br[i] = *(const i4*)(wq + (size_t)(nBase + r) * D_IN + kn + c);
            }
        }

        // Fragment loads follow the documented wave32 VGPR layouts:
        // A 16x32 f16: lanes 0-15 -> K {0..7,16..23}, lanes 16-31 -> K {8..15,24..31}
        v16h af[4];
        #pragma unroll
        for (int mt = 0; mt < 4; mt++) {
            const _Float16* ap = &As[buf][wm * 64 + mt * 16 + ln][lhi * 8];
            ABFrag a;
            a.h[0] = *(const h8*)(ap);
            a.h[1] = *(const h8*)(ap + 16);
            af[mt] = a.v;
        }
        // B 32x16 f16: lane = N, lanes 0-15 -> K {0..7,16..23}, lanes 16-31 -> K {8..15,24..31}
        v16h bfr[4];
        #pragma unroll
        for (int nt = 0; nt < 4; nt++) {
            const _Float16* bp = &Bs[buf][wn * 64 + nt * 16 + ln][lhi * 8];
            ABFrag b;
            b.h[0] = *(const h8*)(bp);
            b.h[1] = *(const h8*)(bp + 16);
            bfr[nt] = b.v;
        }

        #pragma unroll
        for (int mt = 0; mt < 4; mt++)
            #pragma unroll
            for (int nt = 0; nt < 4; nt++)
                acc[mt][nt] = __builtin_amdgcn_wmma_f32_16x16x32_f16(
                    false, af[mt], false, bfr[nt],
                    (short)0, acc[mt][nt], false, false);

        if (pf) {
            const int nb = buf ^ 1;
            #pragma unroll
            for (int i = 0; i < 4; i++) {
                int idx = tid + i * 256;
                int r = idx >> 3;
                int c = (idx & 7) << 2;
                *(h4*)(&As[nb][r][c]) = cvt_f4_h4(ar[i]);
            }
            #pragma unroll
            for (int i = 0; i < 8; i++) {
                int idx = tid + i * 256;
                int r = idx >> 3;
                int c = (idx & 7) << 2;
                *(h4*)(&Bs[nb][r][c]) = cvt_i4_h4(br[i]);
            }
        }
        __syncthreads();
    }

    // ---- epilogue: apply per-output scale + bias, nontemporal scatter ----
    // C/D 16x16 f32 layout: VGPR j -> M = j (lanes 0-15) or 8+j (lanes 16-31), N = lane%16
    #pragma unroll
    for (int nt = 0; nt < 4; nt++) {
        const int col = nBase + wn * 64 + nt * 16 + ln;
        const float s = scale[col];
        const float b = bias[col];
        #pragma unroll
        for (int mt = 0; mt < 4; mt++) {
            const int row0 = mBase + wm * 64 + mt * 16 + lhi * 8;
            v8f a = acc[mt][nt];
            #pragma unroll
            for (int j = 0; j < 8; j++) {
                __builtin_nontemporal_store(a[j] * s + b,
                    out + (size_t)(row0 + j) * D_OUT + col);
            }
        }
    }
}

extern "C" void kernel_launch(void* const* d_in, const int* in_sizes, int n_in,
                              void* d_out, int out_size, void* d_ws, size_t ws_size,
                              hipStream_t stream) {
    (void)in_sizes; (void)n_in; (void)out_size; (void)d_ws; (void)ws_size;
    const float* x     = (const float*)d_in[0];
    const int*   wq    = (const int*)  d_in[1];
    const float* mv    = (const float*)d_in[2];
    const float* bias  = (const float*)d_in[3];
    float*       out   = (float*)d_out;

    const int nBlocks = (TOKENS / BM) * (D_OUT / BN);   // 64 * 43 = 2752
    w8linear_wmma<<<nBlocks, 256, 0, stream>>>(x, wq, mv, bias, out);
}
